// NNUE_14499809591732
// MI455X (gfx1250) — matrix operand order repaired and checked by reading, not compiled, not run on hardware
//
#include <hip/hip_runtime.h>
#include <cstdint>

typedef __bf16 bf16_t;
typedef __attribute__((ext_vector_type(16))) __bf16 v16bf;
typedef __attribute__((ext_vector_type(8)))  __bf16 v8bf;
typedef __attribute__((ext_vector_type(4)))  __bf16 v4bf;
typedef __attribute__((ext_vector_type(8)))  float  v8f;
typedef __attribute__((ext_vector_type(4)))  float  v4f;

#define K_IN   41024
#define B_ROWS 4096
#define L1DIM  256

// ---------------------------------------------------------------------------
// Kernel A: f32 -> bf16 weight conversion (W_in once per launch; W1 likewise)
// ---------------------------------------------------------------------------
__global__ __launch_bounds__(256)
void cvt_f32_bf16(const float* __restrict__ src, bf16_t* __restrict__ dst, int n)
{
  int i = (blockIdx.x * 256 + threadIdx.x) * 8;
  if (i + 7 < n) {
    v4f a = *(const v4f*)(src + i);
    v4f b = *(const v4f*)(src + i + 4);
    v8bf o;
    o[0] = (bf16_t)a.x; o[1] = (bf16_t)a.y; o[2] = (bf16_t)a.z; o[3] = (bf16_t)a.w;
    o[4] = (bf16_t)b.x; o[5] = (bf16_t)b.y; o[6] = (bf16_t)b.z; o[7] = (bf16_t)b.w;
    *(v8bf*)(dst + i) = o;
  }
}

// ---------------------------------------------------------------------------
// Kernel B: feature-transform GEMM.
//   grid (32 mtiles, 2 nhalf, 4 ksplit), 256 threads (8 waves).
//   WG tile: 128 batch rows x 128 cols, BOTH w_in and b_in (shared W tile).
//   Wave tile: 4x4 WMMA tiles (64 rows x 64 cols) -> 512B LDS read / WMMA.
//   A streams (read-once, 1.34 GB): non-temporal loads -> cvt bf16 -> LDS.
//   W tile (bf16, L2-resident): GLOBAL_LOAD_ASYNC_TO_LDS_B128 (ASYNCcnt).
//   Partial sums -> part[split][mat][4096][256] f32 (unique region per WG).
// ---------------------------------------------------------------------------
__global__ __launch_bounds__(256)
void ft_gemm(const float* __restrict__ w_in, const float* __restrict__ b_in,
             const bf16_t* __restrict__ Wb, float* __restrict__ part)
{
  // per-buffer LDS: Aw 128x40 bf16 (10240B) | Ab 10240B | W 128x40 bf16 (10240B)
  __shared__ __align__(16) unsigned char smem[2 * 30720];

  const int tid   = threadIdx.x;
  const int m0    = blockIdx.x * 128;
  const int nhalf = blockIdx.y;          // cols [nhalf*128, +128)
  const int split = blockIdx.z;          // K range

  const int lane  = tid & 31;
  const int wv    = tid >> 5;
  const int lhalf = lane >> 4;
  const int l16   = lane & 15;
  const int rg    = wv & 3;              // row-group (4 row-tiles each)
  const int cg    = wv >> 2;             // col-group (64 cols each)
  const int mat   = rg >> 1;             // 0 = w_in, 1 = b_in
  const int m16b  = (rg & 1) * 4;        // first 16-row tile of this wave

  v8f acc[4][4] = {};

  // 41024/32 = 1282 iterations, split [321,321,320,320]
  const int itStart = split * 320 + (split < 2 ? split : 2);
  const int itCount = 320 + (split < 2 ? 1 : 0);

  auto stage = [&](int buf, int it) {
    unsigned char* base = smem + buf * 30720;
    bf16_t* Aw = (bf16_t*)base;
    bf16_t* Ab = (bf16_t*)(base + 10240);
    bf16_t* Wl = (bf16_t*)(base + 20480);
    const size_t k0 = (size_t)it * 32;
    // A tiles: 128 rows x 32 K f32, both matrices; NT load, cvt, 16B ds store
#pragma unroll
    for (int j = 0; j < 2; ++j) {
      int f = tid + j * 256;                 // 512 8-float chunks per matrix
      int row = f >> 2, col = (f & 3) * 8;
      size_t g = (size_t)(m0 + row) * K_IN + k0 + col;
      v4f a0 = __builtin_nontemporal_load((const v4f*)(w_in + g));
      v4f a1 = __builtin_nontemporal_load((const v4f*)(w_in + g + 4));
      v4f b0 = __builtin_nontemporal_load((const v4f*)(b_in + g));
      v4f b1 = __builtin_nontemporal_load((const v4f*)(b_in + g + 4));
      v8bf av, bv;
      av[0] = (bf16_t)a0.x; av[1] = (bf16_t)a0.y; av[2] = (bf16_t)a0.z; av[3] = (bf16_t)a0.w;
      av[4] = (bf16_t)a1.x; av[5] = (bf16_t)a1.y; av[6] = (bf16_t)a1.z; av[7] = (bf16_t)a1.w;
      bv[0] = (bf16_t)b0.x; bv[1] = (bf16_t)b0.y; bv[2] = (bf16_t)b0.z; bv[3] = (bf16_t)b0.w;
      bv[4] = (bf16_t)b1.x; bv[5] = (bf16_t)b1.y; bv[6] = (bf16_t)b1.z; bv[7] = (bf16_t)b1.w;
      *(v8bf*)(Aw + row * 40 + col) = av;
      *(v8bf*)(Ab + row * 40 + col) = bv;
    }
    // W tile: 128 N-rows x 32 K bf16, DMA'd straight into LDS (no VGPR pass)
#pragma unroll
    for (int j = 0; j < 2; ++j) {
      int c = tid + j * 256;                 // 512 v8bf chunks
      int row = c >> 2, seg = (c & 3) * 8;
      size_t g = (size_t)(nhalf * 128 + row) * K_IN + k0 + seg;
      unsigned long long gaddr =
          (unsigned long long)(uintptr_t)(const void*)(Wb + g);
      unsigned ldsoff =
          (unsigned)(uintptr_t)(const void*)(Wl + row * 40 + seg);
      asm volatile("global_load_async_to_lds_b128 %0, %1, off"
                   :: "v"(ldsoff), "v"(gaddr)
                   : "memory");
    }
  };

  stage(0, itStart);
  for (int i = 0; i < itCount; ++i) {
    asm volatile("s_wait_asynccnt 0x0" ::: "memory");   // LDS-DMA complete
    __syncthreads();
    if (i + 1 < itCount) stage((i + 1) & 1, itStart + i + 1);

    unsigned char* base = smem + (i & 1) * 30720;
    const bf16_t* A  = (const bf16_t*)(base + mat * 10240);
    const bf16_t* Wl = (const bf16_t*)(base + 20480);

    union { v16bf v; v8bf h[2]; } af[4], bfm[4];
    const int ao = lhalf ? 8 : 0;        // A: K {0..7,16..23} / {8..15,24..31}
#pragma unroll
    for (int t = 0; t < 4; ++t) {
      const bf16_t* ar = A + ((m16b + t) * 16 + l16) * 40 + ao;
      af[t].h[0] = *(const v8bf*)(ar);
      af[t].h[1] = *(const v8bf*)(ar + 16);
    }
    const int bo_ = lhalf ? 8 : 0;       // B: lane = N column, K {0..7,16..23} / {8..15,24..31}
#pragma unroll
    for (int n = 0; n < 4; ++n) {
      const bf16_t* br = Wl + ((cg * 4 + n) * 16 + l16) * 40 + bo_;
      bfm[n].h[0] = *(const v8bf*)(br);
      bfm[n].h[1] = *(const v8bf*)(br + 16);
    }
#pragma unroll
    for (int t = 0; t < 4; ++t)
#pragma unroll
      for (int n = 0; n < 4; ++n)
        acc[t][n] = __builtin_amdgcn_wmma_f32_16x16x32_bf16(
            false, af[t].v, false, bfm[n].v, (short)0, acc[t][n], false, false);
  }

  // write partial sums (each (split,mat,row,col) owned by exactly one WG)
  const size_t ms = (size_t)B_ROWS * L1DIM;
  float* pb = part + (size_t)(split * 2 + mat) * ms;
#pragma unroll
  for (int t = 0; t < 4; ++t) {
    int rowb = m0 + (m16b + t) * 16 + 8 * lhalf;   // D: row = r + 8*(lane/16)
#pragma unroll
    for (int n = 0; n < 4; ++n) {
      int col = nhalf * 128 + (cg * 4 + n) * 16 + l16;
#pragma unroll
      for (int r = 0; r < 8; ++r)
        pb[(size_t)(rowb + r) * L1DIM + col] = acc[t][n][r];
    }
  }
}

// ---------------------------------------------------------------------------
// Kernel B2: reduce K-split partials, add bias, us/them mix, clamp -> l0 bf16
// ---------------------------------------------------------------------------
__global__ __launch_bounds__(256)
void reduce_mix(const float* __restrict__ part, const float* __restrict__ bias,
                const float* __restrict__ us, const float* __restrict__ them,
                bf16_t* __restrict__ l0)
{
  int idx = blockIdx.x * 256 + threadIdx.x;      // 4096*64
  int row = idx >> 6;
  int cg  = (idx & 63) * 4;
  const size_t ms  = (size_t)B_ROWS * L1DIM;
  const size_t off = (size_t)row * L1DIM + cg;
  float u = us[row], th = them[row];

  v4f w = *(const v4f*)(part + off);
  v4f b = *(const v4f*)(part + ms + off);
#pragma unroll
  for (int s = 1; s < 4; ++s) {
    w += *(const v4f*)(part + (size_t)(2 * s) * ms + off);
    b += *(const v4f*)(part + (size_t)(2 * s + 1) * ms + off);
  }
  v4f bs = *(const v4f*)(bias + cg);
  w += bs; b += bs;

  v4bf o1, o2;
#pragma unroll
  for (int q = 0; q < 4; ++q) {
    float a = u * w[q] + th * b[q];
    float c = u * b[q] + th * w[q];
    a = a < 0.f ? 0.f : (a > 1.f ? 1.f : a);
    c = c < 0.f ? 0.f : (c > 1.f ? 1.f : c);
    o1[q] = (bf16_t)a; o2[q] = (bf16_t)c;
  }
  *(v4bf*)(l0 + (size_t)row * 512 + cg) = o1;
  *(v4bf*)(l0 + (size_t)row * 512 + 256 + cg) = o2;
}

// ---------------------------------------------------------------------------
// Kernel C: tail. l1 = clip(l0 @ W1^T + b1) via WMMA (K=512, N=32),
//           then per-row scalar l2 = clip(W2 l1 + b2), out = Wo.l2 + bo.
//   grid 64 WGs x 128 threads (4 waves, 16 rows each).
// ---------------------------------------------------------------------------
__global__ __launch_bounds__(128)
void nnue_tail(const bf16_t* __restrict__ l0, const bf16_t* __restrict__ W1b,
               const float* __restrict__ b1, const float* __restrict__ W2,
               const float* __restrict__ b2, const float* __restrict__ Wo,
               const float* __restrict__ bo, float* __restrict__ out)
{
  __shared__ __align__(16) bf16_t w1l[32 * 520];
  __shared__ float l1f[64 * 33];
  __shared__ float w2l[32 * 32];
  __shared__ float b1l[32], b2l[32], wol[32];

  const int tid = threadIdx.x;
#pragma unroll
  for (int j = 0; j < 16; ++j) {               // W1: 2048 v8bf chunks
    int c = tid + j * 128;
    int row = c >> 6, seg = (c & 63) * 8;
    *(v8bf*)(w1l + row * 520 + seg) = *(const v8bf*)(W1b + row * 512 + seg);
  }
#pragma unroll
  for (int j = 0; j < 8; ++j) w2l[tid + j * 128] = W2[tid + j * 128];
  if (tid < 32) { b1l[tid] = b1[tid]; b2l[tid] = b2[tid]; wol[tid] = Wo[tid]; }
  __syncthreads();

  const int wave = tid >> 5, lane = tid & 31;
  const int lhalf = lane >> 4, l16 = lane & 15;
  const size_t mbase = (size_t)blockIdx.x * 64 + wave * 16;

  v8f acc[2] = {};
  const bf16_t* arow = l0 + (mbase + l16) * 512;   // A frags direct from global
#pragma unroll 4
  for (int kk = 0; kk < 512; kk += 32) {
    union { v16bf v; v8bf h[2]; } af;
    int ao = kk + (lhalf ? 8 : 0);
    af.h[0] = *(const v8bf*)(arow + ao);
    af.h[1] = *(const v8bf*)(arow + ao + 16);
#pragma unroll
    for (int ns = 0; ns < 2; ++ns) {
      const bf16_t* wr = w1l + (ns * 16 + l16) * 520 + kk + (lhalf ? 8 : 0);
      union { v16bf v; v8bf h[2]; } bfm;
      bfm.h[0] = *(const v8bf*)(wr);
      bfm.h[1] = *(const v8bf*)(wr + 16);
      acc[ns] = __builtin_amdgcn_wmma_f32_16x16x32_bf16(
          false, af.v, false, bfm.v, (short)0, acc[ns], false, false);
    }
  }
  {
    int row0 = wave * 16 + 8 * lhalf;
#pragma unroll
    for (int ns = 0; ns < 2; ++ns) {
      int col = ns * 16 + l16;
      float bb = b1l[col];
#pragma unroll
      for (int r = 0; r < 8; ++r) {
        float v = acc[ns][r] + bb;
        v = v < 0.f ? 0.f : (v > 1.f ? 1.f : v);
        l1f[(row0 + r) * 33 + col] = v;
      }
    }
  }
  __syncthreads();
  if (tid < 64) {
    const float* l1p = l1f + tid * 33;
    float o = bo[0];
#pragma unroll 4
    for (int j = 0; j < 32; ++j) {
      float s = b2l[j];
#pragma unroll
      for (int i = 0; i < 32; ++i) s += w2l[j * 32 + i] * l1p[i];
      s = s < 0.f ? 0.f : (s > 1.f ? 1.f : s);
      o += wol[j] * s;
    }
    out[(size_t)blockIdx.x * 64 + tid] = o;
  }
}

// ---------------------------------------------------------------------------
extern "C" void kernel_launch(void* const* d_in, const int* in_sizes, int n_in,
                              void* d_out, int out_size, void* d_ws, size_t ws_size,
                              hipStream_t stream)
{
  (void)in_sizes; (void)n_in; (void)out_size; (void)ws_size;
  const float* us   = (const float*)d_in[0];
  const float* them = (const float*)d_in[1];
  const float* w_in = (const float*)d_in[2];
  const float* b_in = (const float*)d_in[3];
  const float* W_in = (const float*)d_in[4];
  const float* bias = (const float*)d_in[5];
  const float* W1   = (const float*)d_in[6];
  const float* b1   = (const float*)d_in[7];
  const float* W2   = (const float*)d_in[8];
  const float* b2   = (const float*)d_in[9];
  const float* Wo   = (const float*)d_in[10];
  const float* bo   = (const float*)d_in[11];
  float* out = (float*)d_out;

  // workspace layout
  bf16_t* Wb   = (bf16_t*)d_ws;                  // 256*41024 bf16  (21.0 MB)
  bf16_t* W1b  = Wb + (size_t)256 * K_IN;        // 32*512 bf16
  bf16_t* l0   = W1b + 32 * 512;                 // 4096*512 bf16   (4.2 MB)
  float*  part = (float*)(l0 + (size_t)B_ROWS * 512);  // 8*4096*256 f32 (33.5 MB)

  const int nW = 256 * K_IN;
  cvt_f32_bf16<<<(nW / 8 + 255) / 256, 256, 0, stream>>>(W_in, Wb, nW);
  cvt_f32_bf16<<<(16384 / 8 + 255) / 256, 256, 0, stream>>>(W1, W1b, 16384);

  ft_gemm<<<dim3(32, 2, 4), 256, 0, stream>>>(w_in, b_in, Wb, part);
  reduce_mix<<<(B_ROWS * 64) / 256, 256, 0, stream>>>(part, bias, us, them, l0);
  nnue_tail<<<B_ROWS / 64, 128, 0, stream>>>(l0, W1b, b1, W2, b2, Wo, bo, out);
}
